// Model_bk_31885837205645
// MI455X (gfx1250) — matrix operand order repaired and checked by reading, not compiled, not run on hardware
//
#include <hip/hip_runtime.h>

typedef __attribute__((ext_vector_type(16))) _Float16 v16h_t;
typedef __attribute__((ext_vector_type(8)))  _Float16 v8h_t;
typedef __attribute__((ext_vector_type(4)))  _Float16 v4h_t;
typedef __attribute__((ext_vector_type(8)))  float    v8f_t;
typedef __attribute__((ext_vector_type(4)))  float    v4f_t;

#define KCHUNK 1024
#define ASTRIDE (KCHUNK + 8)     // pad LDS row stride to dodge bank conflicts
#define CONV_THREADS 128

// ---------------- weight pack: W(27,cin,cout) f32 -> Wt[cout][Kpad] f16 ----
__global__ __launch_bounds__(256)
void pack_w_kernel(const float* __restrict__ W, _Float16* __restrict__ Wt,
                   int cin, int cout, int Kpad) {
  int i = blockIdx.x * blockDim.x + threadIdx.x;
  int total = cout * Kpad;
  if (i >= total) return;
  int d = i / Kpad;
  int k = i - d * Kpad;
  int Ktot = 27 * cin;
  float v = (k < Ktot) ? W[(size_t)k * cout + d] : 0.0f;
  Wt[i] = (_Float16)v;
}

// ---------------- gather-GEMM-scatter conv + BN + ReLU via WMMA ------------
__global__ __launch_bounds__(CONV_THREADS)
void conv_wmma_kernel(const float* __restrict__ feat, int N_in, int cin,
                      const int* __restrict__ nbr, int N_out,
                      const _Float16* __restrict__ Wt, int Kpad,
                      const float* __restrict__ scale,
                      const float* __restrict__ bias,
                      float* __restrict__ out, int cout) {
  __shared__ _Float16 sA[16 * ASTRIDE];
  __shared__ int sIdx[16 * 27];

  const int tid  = threadIdx.x;
  const int lane = tid & 31;
  const int wave = tid >> 5;
  const int rbase = blockIdx.x << 4;

  // neighbor indices for the 16 output voxels (sentinel N_in for OOB rows)
  for (int i = tid; i < 16 * 27; i += CONV_THREADS) {
    int m = i / 27;
    int t = i - m * 27;
    int row = rbase + m;
    sIdx[i] = (row < N_out) ? nbr[(size_t)row * 27 + t] : N_in;
  }
  __syncthreads();

  const int Ktot  = 27 * cin;
  const int ntile = cout >> 4;
  const int nb = lane & 15;
  const int hA = (lane >> 4) << 4;   // A K-half offset: 0 / 16
  const int hB = (lane >> 4) << 4;   // B K-half offset: 0 / 16
  const int t0 = wave;
  const int t1 = wave + 4;

  v8f_t acc0 = {};
  v8f_t acc1 = {};

  for (int kbase = 0; kbase < Kpad; kbase += KCHUNK) {
    const int kc = (Kpad - kbase) < KCHUNK ? (Kpad - kbase) : KCHUNK;

    if ((cin & 3) == 0) {
      // zero the <=31-element K pad tail (last chunk only)
      if (kbase + kc > Ktot) {
        int z0 = (Ktot > kbase) ? (Ktot - kbase) : 0;
        for (int i = tid; i < 16 * 32; i += CONV_THREADS) {
          int m = i >> 5;
          int k = z0 + (i & 31);
          if (k < kc) sA[m * ASTRIDE + k] = (_Float16)0.0f;
        }
      }
      // vectorized gather: 4 threads per (voxel m, tap t) pair, float4 copies
      const int sub = (tid & 3) << 2;
      for (int pr = tid >> 2; pr < 16 * 27; pr += CONV_THREADS / 4) {
        int m = pr / 27;                 // constant divisor
        int t = pr - m * 27;
        int kstart = t * cin;
        int lo = kstart > kbase ? kstart : kbase;
        int hiA = kstart + cin;
        int hiB = kbase + kc;
        int hi = hiA < hiB ? hiA : hiB;
        if (lo >= hi) continue;
        int idx = sIdx[pr];
        bool valid = (unsigned)idx < (unsigned)N_in;
        const float* src = feat + (size_t)idx * cin;
        for (int kk = lo + sub; kk < hi; kk += 16) {
          v4f_t f = {0.0f, 0.0f, 0.0f, 0.0f};
          if (valid) f = *(const v4f_t*)(src + (kk - kstart));
          v4h_t h;
          h[0] = (_Float16)f[0]; h[1] = (_Float16)f[1];
          h[2] = (_Float16)f[2]; h[3] = (_Float16)f[3];
          *(v4h_t*)&sA[m * ASTRIDE + (kk - kbase)] = h;
        }
      }
    } else {
      // scalar fallback (stem conv with cin == 3)
      for (int i = tid; i < 16 * kc; i += CONV_THREADS) {
        int m = i / kc;
        int k = i - m * kc;
        int kg = kbase + k;
        _Float16 v = (_Float16)0.0f;
        if (kg < Ktot) {
          int t = kg / cin;
          int c = kg - t * cin;
          int idx = sIdx[m * 27 + t];
          if ((unsigned)idx < (unsigned)N_in)
            v = (_Float16)feat[(size_t)idx * cin + c];
        }
        sA[m * ASTRIDE + k] = v;
      }
    }
    __syncthreads();

    for (int k = 0; k < kc; k += 32) {
      // A fragment (contiguous K order, same as B: lanes<16 hold K 0..15, lanes>=16 hold K 16..31)
      v8h_t a0 = *(const v8h_t*)&sA[nb * ASTRIDE + k + hA];
      v8h_t a1 = *(const v8h_t*)&sA[nb * ASTRIDE + k + hA + 8];
      v16h_t av;
      #pragma unroll
      for (int j = 0; j < 8; ++j) { av[j] = a0[j]; av[j + 8] = a1[j]; }

      const int kg = kbase + k;
      if (t0 < ntile) {   // wave-uniform branch: EXEC stays all-1s
        v16h_t bv = *(const v16h_t*)(Wt + (size_t)((t0 << 4) + nb) * Kpad + kg + hB);
        acc0 = __builtin_amdgcn_wmma_f32_16x16x32_f16(
                 false, av, false, bv, (short)0, acc0, false, false);
      }
      if (t1 < ntile) {
        v16h_t bv = *(const v16h_t*)(Wt + (size_t)((t1 << 4) + nb) * Kpad + kg + hB);
        acc1 = __builtin_amdgcn_wmma_f32_16x16x32_f16(
                 false, av, false, bv, (short)0, acc1, false, false);
      }
    }
    __syncthreads();
  }

  // epilogue: C/D layout -> VGPR r is M=r (lanes<16) or M=r+8 (lanes>=16)
  const int mb = (lane < 16) ? 0 : 8;
  if (t0 < ntile) {
    int n = (t0 << 4) + nb;
    float sc = scale[n], bi = bias[n];
    #pragma unroll
    for (int r = 0; r < 8; ++r) {
      int m = rbase + mb + r;
      if (m < N_out) {
        float v = acc0[r] * sc + bi;
        out[(size_t)m * cout + n] = v > 0.0f ? v : 0.0f;
      }
    }
  }
  if (t1 < ntile) {
    int n = (t1 << 4) + nb;
    float sc = scale[n], bi = bias[n];
    #pragma unroll
    for (int r = 0; r < 8; ++r) {
      int m = rbase + mb + r;
      if (m < N_out) {
        float v = acc1[r] * sc + bi;
        out[(size_t)m * cout + n] = v > 0.0f ? v : 0.0f;
      }
    }
  }
}

// ---------------- channel concat for skip connections ----------------------
__global__ __launch_bounds__(256)
void concat_kernel(const float* __restrict__ a, const float* __restrict__ b,
                   float* __restrict__ o, int N, int C) {
  size_t i = (size_t)blockIdx.x * blockDim.x + threadIdx.x;
  size_t total = (size_t)N * 2 * C;
  if (i >= total) return;
  int twoC = 2 * C;
  size_t n = i / twoC;
  int c = (int)(i - n * twoC);
  o[i] = (c < C) ? a[n * C + c] : b[n * C + (c - C)];
}

// ---------------- final linear: (N,16) @ (16,20) + b -----------------------
__global__ __launch_bounds__(256)
void linear_kernel(const float* __restrict__ f, const float* __restrict__ W,
                   const float* __restrict__ b, float* __restrict__ out, int N) {
  size_t i = (size_t)blockIdx.x * blockDim.x + threadIdx.x;
  if (i >= (size_t)N * 20) return;
  size_t n = i / 20;
  int j = (int)(i - n * 20);
  float s = b[j];
  #pragma unroll
  for (int c = 0; c < 16; ++c) s += f[n * 16 + c] * W[c * 20 + j];
  out[i] = s;
}

extern "C" void kernel_launch(void* const* d_in, const int* in_sizes, int n_in,
                              void* d_out, int out_size, void* d_ws, size_t ws_size,
                              hipStream_t stream) {
  (void)n_in; (void)out_size; (void)ws_size;
  int p = 0;
  const float* feats = (const float*)d_in[p];
  const int N0 = in_sizes[p] / 3; ++p;

  struct Blk { const float *W, *scale, *bias; int cin, cout; };
  auto take = [&](int cin, int cout) {
    Blk b; b.W = (const float*)d_in[p++]; b.scale = (const float*)d_in[p++];
    b.bias = (const float*)d_in[p++]; b.cin = cin; b.cout = cout; return b;
  };
  Blk stem[3]; stem[0] = take(3, 16); stem[1] = take(16, 16); stem[2] = take(16, 16);
  Blk down[6][3];
  for (int i = 0; i < 6; ++i) {
    down[i][0] = take((i + 1) * 16, (i + 2) * 16);
    down[i][1] = take((i + 2) * 16, (i + 2) * 16);
    down[i][2] = take((i + 2) * 16, (i + 2) * 16);
  }
  Blk upb[6];
  for (int i = 0; i < 6; ++i) upb[i] = take((i + 2) * 16, (i + 1) * 16);
  Blk fuse[6][2];
  for (int i = 0; i < 6; ++i) {
    fuse[i][0] = take(2 * (i + 1) * 16, (i + 1) * 16);
    fuse[i][1] = take((i + 1) * 16, (i + 1) * 16);
  }
  const float* linW = (const float*)d_in[p++];
  const float* linB = (const float*)d_in[p++];

  const int* subm[7]; int Nl[7];
  for (int l = 0; l < 7; ++l) { subm[l] = (const int*)d_in[p]; Nl[l] = in_sizes[p] / 27; ++p; }
  const int* dnb[6]; for (int i = 0; i < 6; ++i) dnb[i] = (const int*)d_in[p++];
  const int* unb[6]; for (int i = 0; i < 6; ++i) unb[i] = (const int*)d_in[p++];
  (void)N0;

  // ---- workspace carve-out (host-side arithmetic only; capture-safe) ----
  char* base = (char*)d_ws;
  size_t off = 0;
  auto alloc = [&](size_t bytes) -> void* {
    void* r = base + off;
    off += (bytes + 255) & ~(size_t)255;
    return r;
  };
  float* xs[7];
  for (int l = 0; l < 7; ++l)
    xs[l] = (float*)alloc((size_t)Nl[l] * (l + 1) * 16 * sizeof(float));
  size_t maxT = 0;
  for (int l = 0; l < 7; ++l) {
    size_t s = (size_t)Nl[l] * 2 * (l + 1) * 16;
    if (s > maxT) maxT = s;
  }
  float* T0 = (float*)alloc(maxT * sizeof(float));
  float* T1 = (float*)alloc(maxT * sizeof(float));
  size_t maxW = 0;
  auto wsz = [&](const Blk& b) {
    size_t kp = ((size_t)27 * b.cin + 31) & ~(size_t)31;
    size_t s = kp * b.cout;
    if (s > maxW) maxW = s;
  };
  for (int i = 0; i < 3; ++i) wsz(stem[i]);
  for (int i = 0; i < 6; ++i) {
    wsz(down[i][0]); wsz(down[i][1]); wsz(down[i][2]);
    wsz(upb[i]); wsz(fuse[i][0]); wsz(fuse[i][1]);
  }
  _Float16* Wt = (_Float16*)alloc(maxW * sizeof(_Float16));

  auto conv = [&](const Blk& b, const float* fin, int Nin, const int* nb,
                  int Nout, float* fout) {
    int Kpad = (27 * b.cin + 31) & ~31;
    int total = b.cout * Kpad;
    pack_w_kernel<<<(total + 255) / 256, 256, 0, stream>>>(b.W, Wt, b.cin, b.cout, Kpad);
    conv_wmma_kernel<<<dim3((Nout + 15) / 16), CONV_THREADS, 0, stream>>>(
        fin, Nin, b.cin, nb, Nout, Wt, Kpad, b.scale, b.bias, fout, b.cout);
  };

  // stem (level 0)
  conv(stem[0], feats, Nl[0], subm[0], Nl[0], T0);
  conv(stem[1], T0, Nl[0], subm[0], Nl[0], T1);
  conv(stem[2], T1, Nl[0], subm[0], Nl[0], xs[0]);

  // encoder
  for (int i = 0; i < 6; ++i) {
    conv(down[i][0], xs[i], Nl[i], dnb[i], Nl[i + 1], T0);       // stride-2 conv
    conv(down[i][1], T0, Nl[i + 1], subm[i + 1], Nl[i + 1], T1); // subm conv
    conv(down[i][2], T1, Nl[i + 1], subm[i + 1], Nl[i + 1], xs[i + 1]);
  }

  // decoder
  float* R = xs[6];
  for (int i = 5; i >= 0; --i) {
    int Cl = (i + 1) * 16;
    conv(upb[i], R, Nl[i + 1], unb[i], Nl[i], T0);               // inverse conv
    size_t tot = (size_t)Nl[i] * 2 * Cl;
    concat_kernel<<<(unsigned)((tot + 255) / 256), 256, 0, stream>>>(
        T0, xs[i], T1, Nl[i], Cl);                               // skip concat
    conv(fuse[i][0], T1, Nl[i], subm[i], Nl[i], T0);
    conv(fuse[i][1], T0, Nl[i], subm[i], Nl[i], T1);
    R = T1;
  }

  // head
  size_t tot = (size_t)Nl[0] * 20;
  linear_kernel<<<(unsigned)((tot + 255) / 256), 256, 0, stream>>>(
      R, linW, linB, (float*)d_out, Nl[0]);
}
